// SimpleRGCN_85701777425175
// MI455X (gfx1250) — matrix operand order repaired and checked by reading, not compiled, not run on hardware
//
#include <hip/hip_runtime.h>
#include <hip/hip_bf16.h>
#include <stdint.h>

// Problem constants (match the reference)
#define N_NODES 100000
#define R_REL   8
#define HFR_DIM 128
#define HTO_DIM 128
#define NNZ_E   1600000
#define K_DIM   (R_REL * HFR_DIM)   // 1024 contracted dim of the fused GEMM

// LDS weight slab: 128 rows x (128 + 8 pad) bf16 -> 272B row stride
// (pad makes B-fragment ds_load_b128 reads bank-conflict-minimal)
#define SW_STRIDE 136

typedef __attribute__((ext_vector_type(16))) __bf16 v16bf;
typedef __attribute__((ext_vector_type(8)))  __bf16 v8bf;
typedef __attribute__((ext_vector_type(8)))  float  v8f;
typedef __attribute__((ext_vector_type(4)))  float  v4f;
typedef __attribute__((ext_vector_type(4)))  unsigned int v4u;

// pack two f32 into a bf16x2 dword (lowers to v_cvt_pk_bf16_f32)
static __device__ __forceinline__ unsigned int pack_bf16x2(float a, float b) {
    union { __bf16 h[2]; unsigned int u; } p;
    p.h[0] = (__bf16)a;
    p.h[1] = (__bf16)b;
    return p.u;
}

// ---------------------------------------------------------------------------
// Kernel 1: zero the bf16 agg workspace (N*R*HFR bf16 = 204.8 MB)
// 12.8M x 16B stores; 50000 blocks x 256 threads covers it exactly.
// ---------------------------------------------------------------------------
__global__ __launch_bounds__(256) void rgcn_zero_agg(v4u* __restrict__ agg) {
    size_t i = (size_t)blockIdx.x * 256 + threadIdx.x;
    v4u z = {0u, 0u, 0u, 0u};
    agg[i] = z;
}

// ---------------------------------------------------------------------------
// Kernel 2: convert weights fp32 -> bf16 (same [r][i][j] layout, 131072 elems)
// ---------------------------------------------------------------------------
__global__ __launch_bounds__(256) void rgcn_cvt_weights(const float* __restrict__ w,
                                                        __bf16* __restrict__ wbf) {
    int i = blockIdx.x * 256 + threadIdx.x;
    wbf[i] = (__bf16)w[i];
}

// ---------------------------------------------------------------------------
// Kernel 3: SpMM  agg[row][0:128] += value * nodes[col][0:128]  (bf16 accum)
// One wave per edge; each lane owns 4 features -> 2 packed-bf16 atomics.
// nodes (51.2MB) is L2-resident; global_atomic_pk_add_bf16 halves scatter
// traffic vs fp32 atomics (avg 2 edges/row -> bf16 accumulation is benign).
// ---------------------------------------------------------------------------
__global__ __launch_bounds__(256) void rgcn_spmm(const float* __restrict__ nodes,
                                                 const long long* __restrict__ rows,
                                                 const long long* __restrict__ cols,
                                                 const float* __restrict__ vals,
                                                 __bf16* __restrict__ agg) {
    int lane = threadIdx.x & 31;
    long long e = (long long)blockIdx.x * 8 + (threadIdx.x >> 5);
    if (e >= NNZ_E) return;

    long long row = rows[e];          // uniform across the wave
    long long col = cols[e];
    float     v   = vals[e];

    v4f x = *(const v4f*)(nodes + (size_t)col * HFR_DIM + lane * 4);

    __bf16* dst = agg + (size_t)row * HFR_DIM + lane * 4;
    unsigned int d0 = pack_bf16x2(x.x * v, x.y * v);
    unsigned int d1 = pack_bf16x2(x.z * v, x.w * v);
    asm volatile("global_atomic_pk_add_bf16 %0, %1, off"
                 :: "v"(dst),     "v"(d0) : "memory");
    asm volatile("global_atomic_pk_add_bf16 %0, %1, off"
                 :: "v"(dst + 2), "v"(d1) : "memory");
}

// ---------------------------------------------------------------------------
// Kernel 4: fused GEMM  out[n,i] = relu( sum_{r,j} W[r,i,j] * agg[r,n,j] )
// M = N_NODES, Nout = 128, K = 1024 (r-major).  Each wave computes a
// 16(M) x 128(N) strip: 8 accumulator tiles, K in chunks of 32 via
// v_wmma_f32_16x16x32_bf16.
//
// Per relation, the 32KB bf16 weight slab is staged once per workgroup into
// LDS with global_load_async_to_lds_b128 (ASYNCcnt), then all 8 waves read
// their B fragments from LDS (ds_load_b128, padded stride -> balanced banks).
// This cuts the weight stream from L2 by 8x (per-block instead of per-wave).
// ---------------------------------------------------------------------------
__global__ __launch_bounds__(256) void rgcn_gemm(const __bf16* __restrict__ agg,
                                                 const __bf16* __restrict__ wbf,
                                                 float* __restrict__ out) {
    __shared__ __bf16 sw[128 * SW_STRIDE];     // 34,816 B

    const int wave    = threadIdx.x >> 5;       // 0..7
    const int lane    = threadIdx.x & 31;
    const int l15     = lane & 15;
    const int lhalf   = lane >> 4;              // 0 or 1
    const int rowTile = blockIdx.x * 8 + wave;  // 16-row tile index
    const int m0      = rowTile * 16;

    // clamped row for A loads (stores are predicated, so clamping is safe)
    int arow_idx = m0 + l15;
    if (arow_idx >= N_NODES) arow_idx = N_NODES - 1;

    v8f acc[8] = {};

    for (int r = 0; r < R_REL; ++r) {
        __syncthreads();   // all waves done reading the previous slab

        // ---- stage relation r's weights (128x128 bf16 = 32KB) into LDS ----
        // 2048 16B chunks / 256 threads = 8 async copies per thread
        for (int c = threadIdx.x; c < 2048; c += 256) {
            int row = c >> 4;
            int col = (c & 15) * 8;            // bf16 elements
            const __bf16* g = wbf + ((size_t)r * HFR_DIM + row) * HTO_DIM + col;
            unsigned lds = (unsigned)(uintptr_t)&sw[row * SW_STRIDE + col];
            asm volatile("global_load_async_to_lds_b128 %0, %1, off"
                         :: "v"(lds), "v"(g) : "memory");
        }
        asm volatile("s_wait_asynccnt 0x0" ::: "memory");
        __syncthreads();   // slab visible to all waves

        // ---- 4 k-chunks of 32 within this relation ----
        for (int jc = 0; jc < 4; ++jc) {
            const int j0 = jc * 32;

            // A fragment: 2 x 8 contiguous bf16 (16B b128 global loads)
            const __bf16* ap = agg + ((size_t)r * N_NODES + arow_idx) * HFR_DIM
                                   + j0 + lhalf * 8;
            union { v16bf v; v8bf h[2]; } af;
            af.h[0] = *(const v8bf*)(ap);
            af.h[1] = *(const v8bf*)(ap + 16);

            // 8 output tiles along the 128 output features, B from LDS
#pragma unroll
            for (int nt = 0; nt < 8; ++nt) {
                const int i0 = nt * 16;
                const __bf16* bp = &sw[(i0 + l15) * SW_STRIDE + j0 + lhalf * 8];
                union { v16bf v; v8bf h[2]; } bf;
                bf.h[0] = *(const v8bf*)(bp);
                bf.h[1] = *(const v8bf*)(bp + 16);
                acc[nt] = __builtin_amdgcn_wmma_f32_16x16x32_bf16(
                    /*neg_a=*/false, af.v, /*neg_b=*/false, bf.v,
                    /*c_mod=*/(short)0, acc[nt],
                    /*reuse_a=*/false, /*reuse_b=*/false);
            }
        }
    }

    // ---- epilogue: relu + store.  C/D layout: VGPR v holds M = m0+v
    // (lanes 0-15) or m0+v+8 (lanes 16-31); column = nt*16 + l15.
    if (m0 + 16 <= N_NODES) {
        // fast path: full tile, no predication
#pragma unroll
        for (int nt = 0; nt < 8; ++nt) {
            union { v8f v; float f[8]; } c;
            c.v = acc[nt];
#pragma unroll
            for (int v = 0; v < 8; ++v) {
                int m = m0 + v + lhalf * 8;
                float y = c.f[v];
                out[(size_t)m * HTO_DIM + nt * 16 + l15] = y > 0.f ? y : 0.f;
            }
        }
    } else if (m0 < N_NODES) {
        // tail tile: per-row predication
#pragma unroll
        for (int nt = 0; nt < 8; ++nt) {
            union { v8f v; float f[8]; } c;
            c.v = acc[nt];
#pragma unroll
            for (int v = 0; v < 8; ++v) {
                int m = m0 + v + lhalf * 8;
                if (m < N_NODES) {
                    float y = c.f[v];
                    out[(size_t)m * HTO_DIM + nt * 16 + l15] = y > 0.f ? y : 0.f;
                }
            }
        }
    }
}

// ---------------------------------------------------------------------------
// Host-side launcher
// ---------------------------------------------------------------------------
extern "C" void kernel_launch(void* const* d_in, const int* in_sizes, int n_in,
                              void* d_out, int out_size, void* d_ws, size_t ws_size,
                              hipStream_t stream) {
    const float*      nodes   = (const float*)d_in[0];
    const long long*  indices = (const long long*)d_in[1];  // [2, NNZ] int64
    const float*      vals    = (const float*)d_in[2];
    const float*      weights = (const float*)d_in[3];
    float*            out     = (float*)d_out;

    const long long* rows = indices;
    const long long* cols = indices + NNZ_E;

    // workspace layout: [bf16 agg : N*R*HFR][bf16 weights : R*HFR*HTO]
    __bf16* agg = (__bf16*)d_ws;
    __bf16* wbf = (__bf16*)((char*)d_ws +
                  (size_t)N_NODES * R_REL * HFR_DIM * sizeof(__bf16));

    // 1) zero the bf16 accumulation buffer:
    //    102.4M bf16 / 8 per thread / 256 = 50000 blocks
    rgcn_zero_agg<<<50000, 256, 0, stream>>>((v4u*)agg);

    // 2) weights fp32 -> bf16 (131072 elems)
    rgcn_cvt_weights<<<(R_REL * HFR_DIM * HTO_DIM) / 256, 256, 0, stream>>>(weights, wbf);

    // 3) SpMM scatter with packed-bf16 atomics: 8 edges per 256-thread block
    rgcn_spmm<<<(NNZ_E + 7) / 8, 256, 0, stream>>>(nodes, rows, cols, vals, agg);

    // 4) fused per-relation GEMM + relu via bf16 WMMA
    //    row tiles = ceil(100000/16) = 6250; 8 waves/block -> 782 blocks
    rgcn_gemm<<<(6250 + 7) / 8, 256, 0, stream>>>(agg, wbf, out);
}
